// NeRFModel_88433376625123
// MI455X (gfx1250) — matrix operand order repaired and checked by reading, not compiled, not run on hardware
//
#include <hip/hip_runtime.h>
#include <hip/hip_bf16.h>
#include <math.h>

// ---------------- problem constants ----------------
#define NPTS   262144
#define MTILE  64           // points (rows) per workgroup
#define NTH    64           // 2 waves of 32; each wave owns 32 rows (2 WMMA M-tiles)
#define AP     320          // activation LDS pitch (f16 elements), covers K=320 skip layer

typedef __attribute__((ext_vector_type(16))) _Float16 v16h;
typedef __attribute__((ext_vector_type(8)))  _Float16 v8h;
typedef __attribute__((ext_vector_type(8)))  float    v8f;

// ---------------- f16 weight workspace layout (element offsets) ----------------
// All transposed: Wt[n][k] row-major with padded K, so B-fragment reads are contiguous.
#define OFF_WIN   0          // [256][64]   (K 63 -> pad 64)
#define OFF_WH0   16384      // h0..h3: [256][256] each
#define OFF_WH4   278528     // [256][320]  (K 319 -> pad 320: 256 h + 63 pe + pad)
#define OFF_WH5   360448     // h5,h6: [256][256]
#define OFF_WOUT  491520     // [272][256]  (N 257 -> pad 272; row 256 = density col)
#define OFF_WR1   561152     // [128][288]  (K 283 -> 256 feat + 27 de + pad 288)
#define OFF_BIN   598016     // [256]
#define OFF_BH    598272     // [7][256]
#define OFF_BOUT  600064     // [272] (257 -> pad)
#define OFF_BR1   600336     // [128]
#define WTOTAL    600464
#define NBIAS     2448       // WTOTAL - OFF_BIN, staged to LDS
// bias offsets inside the LDS bias block
#define SB_BIN    0
#define SB_BH     256
#define SB_BOUT   2048
#define SB_BR1    2320

// ---------------- weight conversion / transpose / pad ----------------
__global__ __launch_bounds__(256) void prep_weights(
    const float* __restrict__ W_in,
    const float* __restrict__ Wh0, const float* __restrict__ Wh1,
    const float* __restrict__ Wh2, const float* __restrict__ Wh3,
    const float* __restrict__ Wh4, const float* __restrict__ Wh5,
    const float* __restrict__ Wh6,
    const float* __restrict__ W_out, const float* __restrict__ W_r1,
    const float* __restrict__ b_in,
    const float* __restrict__ bh0, const float* __restrict__ bh1,
    const float* __restrict__ bh2, const float* __restrict__ bh3,
    const float* __restrict__ bh4, const float* __restrict__ bh5,
    const float* __restrict__ bh6,
    const float* __restrict__ b_out, const float* __restrict__ b_r1,
    _Float16* __restrict__ wt)
{
  const int i = blockIdx.x * blockDim.x + threadIdx.x;
  if (i >= WTOTAL) return;
  const float* Wh[7] = {Wh0,Wh1,Wh2,Wh3,Wh4,Wh5,Wh6};
  const float* bh[7] = {bh0,bh1,bh2,bh3,bh4,bh5,bh6};
  float v = 0.0f;
  if (i < OFF_WH0) {                       // Wt_in [256][64]
    int n = i >> 6, k = i & 63;
    v = (k < 63) ? W_in[k*256 + n] : 0.0f;
  } else if (i < OFF_WH4) {                // h0..h3 [256][256]
    int j = i - OFF_WH0; int L = j >> 16; int r = j & 65535;
    int n = r >> 8, k = r & 255;
    v = Wh[L][k*256 + n];
  } else if (i < OFF_WH5) {                // h4 [256][320]
    int j = i - OFF_WH4; int n = j / 320, k = j % 320;
    v = (k < 319) ? Wh[4][k*256 + n] : 0.0f;   // rows 0..255 = h, 256..318 = pe
  } else if (i < OFF_WOUT) {               // h5,h6 [256][256]
    int j = i - OFF_WH5; int L = 5 + (j >> 16); int r = j & 65535;
    int n = r >> 8, k = r & 255;
    v = Wh[L][k*256 + n];
  } else if (i < OFF_WR1) {                // Wt_out [272][256]
    int j = i - OFF_WOUT; int n = j >> 8, k = j & 255;
    v = (n < 257) ? W_out[k*257 + n] : 0.0f;
  } else if (i < OFF_BIN) {                // Wt_r1 [128][288]
    int j = i - OFF_WR1; int n = j / 288, k = j % 288;
    v = (k < 283) ? W_r1[k*128 + n] : 0.0f;    // rows 0..255 feat, 256..282 de
  } else if (i < OFF_BH) {
    v = b_in[i - OFF_BIN];
  } else if (i < OFF_BOUT) {
    int j = i - OFF_BH; v = bh[j >> 8][j & 255];
  } else if (i < OFF_BR1) {
    int j = i - OFF_BOUT; v = (j < 257) ? b_out[j] : 0.0f;
  } else {
    v = b_r1[i - OFF_BR1];
  }
  wt[i] = (_Float16)v;
}

// ---------------- fused MLP ----------------
// A-fragment (16-bit A 16x32 ISA layout), lane (m = lane&15, half = lane>>4):
//   elems 0..7  = K k0 + half*8 + e ; elems 8..15 = K k0 + 16 + half*8 + (e-8)
// B-fragment from transposed Wt[n][k]: same K order as A-fragment
// C/D: lane<16: N=lane, M=v ; lane>=16: N=lane-16, M=v+8
// Each wave owns 32 rows = 2 M-tiles; one B-fragment load feeds 2 WMMAs.
// acc starts at 0 (inline-C WMMA); bias is added in the epilogue from LDS.

__device__ inline v16h cat8(v8h a, v8h b) {
  return __builtin_shufflevector(a, b, 0,1,2,3,4,5,6,7,8,9,10,11,12,13,14,15);
}

template<int KSTEPS, bool RELU>
__device__ inline void gemm_layer(const _Float16* inL,
                                  const _Float16* __restrict__ w, int kpad,
                                  const _Float16* biasL,     // LDS
                                  _Float16* outL, int nTiles,
                                  int lane, int r0)
{
  const int m    = lane & 15;
  const int half = lane >> 4;
  const _Float16* ar0 = inL + (r0 + m) * AP + half * 8;
  const _Float16* ar1 = inL + (r0 + 16 + m) * AP + half * 8;
  v16h aF0[KSTEPS], aF1[KSTEPS];
#pragma unroll
  for (int s = 0; s < KSTEPS; ++s) {
    aF0[s] = cat8(*(const v8h*)(ar0 + 32*s), *(const v8h*)(ar0 + 32*s + 16));
    aF1[s] = cat8(*(const v8h*)(ar1 + 32*s), *(const v8h*)(ar1 + 32*s + 16));
  }
  const int mrow0 = r0 + (half ? 8 : 0);
  for (int nt = 0; nt < nTiles; ++nt) {
    const int n0 = nt * 16;
    v8f acc0 = {};
    v8f acc1 = {};
    const _Float16* wrow = w + (n0 + m) * kpad + half * 8;
#pragma unroll
    for (int s = 0; s < KSTEPS; ++s) {
      v16h bF = cat8(*(const v8h*)(wrow + 32*s), *(const v8h*)(wrow + 32*s + 16));
      acc0 = __builtin_amdgcn_wmma_f32_16x16x32_f16(
                 false, aF0[s], false, bF, (short)0, acc0, false, false);
      acc1 = __builtin_amdgcn_wmma_f32_16x16x32_f16(
                 false, aF1[s], false, bF, (short)0, acc1, false, false);
    }
    const float bv = (float)biasL[n0 + m];
    _Float16* o = outL + n0 + m;
#pragma unroll
    for (int v = 0; v < 8; ++v) {
      float x0 = acc0[v] + bv;
      float x1 = acc1[v] + bv;
      if (RELU) { x0 = fmaxf(x0, 0.0f); x1 = fmaxf(x1, 0.0f); }
      o[(mrow0 + v) * AP]      = (_Float16)x0;
      o[(mrow0 + 16 + v) * AP] = (_Float16)x1;
    }
  }
}

__global__ __launch_bounds__(NTH) void nerf_fused(
    const float* __restrict__ pos, const float* __restrict__ dirv,
    const _Float16* __restrict__ wt,
    const float* __restrict__ Wr2, const float* __restrict__ br2,
    float* __restrict__ out)
{
  __shared__ __align__(16) _Float16 sA [MTILE * AP];
  __shared__ __align__(16) _Float16 sB [MTILE * AP];
  __shared__ __align__(16) _Float16 sPE[MTILE * 64];
  __shared__ __align__(16) _Float16 sDE[MTILE * 32];
  __shared__ __align__(16) _Float16 sBias[NBIAS];
  __shared__ __align__(16) float    sWr2[384 + 4];   // W_r2 then b_r2

  const int tid  = threadIdx.x;
  const int lane = tid & 31;
  const int wave = tid >> 5;
  const int r0   = wave * 32;              // this wave's 32 rows
  const int row  = r0 + lane;              // per-lane row for encoding/copies
  const int gp   = blockIdx.x * MTILE + row;

  // warm L2/WGP$ for the first weight layer (global_prefetch_b8 path)
  __builtin_prefetch((const void*)(wt + OFF_WIN + tid * 256), 0, 3);

  // stage biases and the small rgb-head tail weights into LDS
  for (int j = tid; j < NBIAS; j += NTH) sBias[j] = wt[OFF_BIN + j];
  for (int j = tid; j < 384;   j += NTH) sWr2[j]  = Wr2[j];
  if (tid < 3) sWr2[384 + tid] = br2[tid];

  // ---- encodings: each lane builds pe + de for its own row ----
  {
    const float x0 = pos[3*gp+0], x1 = pos[3*gp+1], x2 = pos[3*gp+2];
    _Float16* pe = sPE + row * 64;
    _Float16* a  = sA  + row * AP;
    pe[0] = (_Float16)x0; pe[1] = (_Float16)x1; pe[2] = (_Float16)x2;
    for (int f = 0; f < 10; ++f) {
      const float sc = (float)(1 << f);
      const float xs[3] = {x0*sc, x1*sc, x2*sc};
      for (int c = 0; c < 3; ++c) {
        pe[3 + f*6 + c]     = (_Float16)sinf(xs[c]);
        pe[3 + f*6 + 3 + c] = (_Float16)cosf(xs[c]);
      }
    }
    pe[63] = (_Float16)0.0f;
    for (int j = 0; j < 64; ++j) a[j] = pe[j];

    const float d0 = dirv[3*gp+0], d1 = dirv[3*gp+1], d2 = dirv[3*gp+2];
    _Float16* de = sDE + row * 32;
    de[0] = (_Float16)d0; de[1] = (_Float16)d1; de[2] = (_Float16)d2;
    for (int f = 0; f < 4; ++f) {
      const float sc = (float)(1 << f);
      const float xs[3] = {d0*sc, d1*sc, d2*sc};
      for (int c = 0; c < 3; ++c) {
        de[3 + f*6 + c]     = (_Float16)sinf(xs[c]);
        de[3 + f*6 + 3 + c] = (_Float16)cosf(xs[c]);
      }
    }
    for (int j = 27; j < 32; ++j) de[j] = (_Float16)0.0f;
  }
  __syncthreads();

  // input layer: pe(64) -> 256
  gemm_layer<2,  true>(sA, wt + OFF_WIN, 64,  sBias + SB_BIN,      sB, 16, lane, r0);
  __syncthreads();
  gemm_layer<8,  true>(sB, wt + OFF_WH0            , 256, sBias + SB_BH + 0*256, sA, 16, lane, r0);
  __syncthreads();
  gemm_layer<8,  true>(sA, wt + OFF_WH0 + 1*65536  , 256, sBias + SB_BH + 1*256, sB, 16, lane, r0);
  __syncthreads();
  gemm_layer<8,  true>(sB, wt + OFF_WH0 + 2*65536  , 256, sBias + SB_BH + 2*256, sA, 16, lane, r0);
  __syncthreads();
  gemm_layer<8,  true>(sA, wt + OFF_WH0 + 3*65536  , 256, sBias + SB_BH + 3*256, sB, 16, lane, r0);
  __syncthreads();

  // skip concat: append pe to sB cols 256..319 (one full row per lane)
  {
    const _Float16* pe = sPE + row * 64;
    _Float16* dst = sB + row * AP + 256;
#pragma unroll
    for (int j = 0; j < 64; ++j) dst[j] = pe[j];
  }
  __syncthreads();

  gemm_layer<10, true>(sB, wt + OFF_WH4, 320, sBias + SB_BH + 4*256, sA, 16, lane, r0);
  __syncthreads();
  gemm_layer<8,  true>(sA, wt + OFF_WH5            , 256, sBias + SB_BH + 5*256, sB, 16, lane, r0);
  __syncthreads();
  gemm_layer<8,  true>(sB, wt + OFF_WH5 + 1*65536  , 256, sBias + SB_BH + 6*256, sA, 16, lane, r0);
  __syncthreads();

  // output layer: features (cols 0..255, NO relu) -> sB
  gemm_layer<8, false>(sA, wt + OFF_WOUT, 256, sBias + SB_BOUT, sB, 16, lane, r0);

  // density tile (N-tile 16 of padded output layer): relu, straight to global
  {
    const int m = lane & 15, half = lane >> 4;
    const float bv = (float)sBias[SB_BOUT + 256 + m];
    const _Float16* wrowD = wt + OFF_WOUT + (256 + m) * 256 + half * 8;
#pragma unroll
    for (int mt = 0; mt < 2; ++mt) {
      const _Float16* arow = sA + (r0 + mt*16 + m) * AP + half * 8;
      v8f acc = {};
#pragma unroll
      for (int s = 0; s < 8; ++s) {
        v16h aF = cat8(*(const v8h*)(arow + 32*s), *(const v8h*)(arow + 32*s + 16));
        v16h bF = cat8(*(const v8h*)(wrowD + 32*s), *(const v8h*)(wrowD + 32*s + 16));
        acc = __builtin_amdgcn_wmma_f32_16x16x32_f16(
                  false, aF, false, bF, (short)0, acc, false, false);
      }
      if (m == 0) {                 // these lanes hold column 256 = density
        const int mrow = r0 + mt*16 + (half ? 8 : 0);
#pragma unroll
        for (int v = 0; v < 8; ++v) {
          out[3*NPTS + blockIdx.x * MTILE + mrow + v] = fmaxf(acc[v] + bv, 0.0f);
        }
      }
    }
  }
  __syncthreads();

  // append de to sB cols 256..287 (288..319 zero-weighted anyway)
  {
    const _Float16* de = sDE + row * 32;
    _Float16* dst = sB + row * AP + 256;
#pragma unroll
    for (int j = 0; j < 32; ++j) dst[j] = de[j];
  }
  __syncthreads();

  // rgb head layer 1: (256 feat + 27 de, K=288) -> 128, relu
  gemm_layer<9, true>(sB, wt + OFF_WR1, 288, sBias + SB_BR1, sA, 8, lane, r0);
  __syncthreads();

  // rgb head layer 2: 128 -> 3, sigmoid (one row per lane, weights in LDS)
  {
    float a0 = sWr2[384 + 0], a1 = sWr2[384 + 1], a2 = sWr2[384 + 2];
    const _Float16* h = sA + row * AP;
    for (int j = 0; j < 128; ++j) {
      const float hv = (float)h[j];
      a0 += hv * sWr2[j*3 + 0];
      a1 += hv * sWr2[j*3 + 1];
      a2 += hv * sWr2[j*3 + 2];
    }
    out[3*gp + 0] = 1.0f / (1.0f + expf(-a0));
    out[3*gp + 1] = 1.0f / (1.0f + expf(-a1));
    out[3*gp + 2] = 1.0f / (1.0f + expf(-a2));
  }
}

// ---------------- host entry ----------------
extern "C" void kernel_launch(void* const* d_in, const int* in_sizes, int n_in,
                              void* d_out, int out_size, void* d_ws, size_t ws_size,
                              hipStream_t stream)
{
  (void)in_sizes; (void)n_in; (void)out_size; (void)ws_size;
  const float* pos  = (const float*)d_in[0];
  const float* dirv = (const float*)d_in[1];
  const float* W_in = (const float*)d_in[2];
  const float* b_in = (const float*)d_in[3];
  const float* Wh[7]; const float* bh[7];
  for (int i = 0; i < 7; ++i) {
    Wh[i] = (const float*)d_in[4 + 2*i];
    bh[i] = (const float*)d_in[5 + 2*i];
  }
  const float* W_out = (const float*)d_in[18];
  const float* b_out = (const float*)d_in[19];
  const float* W_r1  = (const float*)d_in[20];
  const float* b_r1  = (const float*)d_in[21];
  const float* W_r2  = (const float*)d_in[22];
  const float* b_r2  = (const float*)d_in[23];

  _Float16* wt = (_Float16*)d_ws;
  float* out = (float*)d_out;

  prep_weights<<<(WTOTAL + 255) / 256, 256, 0, stream>>>(
      W_in, Wh[0], Wh[1], Wh[2], Wh[3], Wh[4], Wh[5], Wh[6],
      W_out, W_r1,
      b_in, bh[0], bh[1], bh[2], bh[3], bh[4], bh[5], bh[6],
      b_out, b_r1, wt);

  nerf_fused<<<NPTS / MTILE, NTH, 0, stream>>>(pos, dirv, wt, W_r2, b_r2, out);
}
